// PGNNConv_21260088115319
// MI455X (gfx1250) — compile-verified, matrix operand order checked by reading, not run on hardware
//
#include <hip/hip_runtime.h>
#include <hip/hip_bf16.h>

#define E_DIM 128
#define K_ANCH 32

typedef __attribute__((ext_vector_type(16))) __bf16 v16bf;
typedef __attribute__((ext_vector_type(8)))  float  v8f;

struct __attribute__((aligned(16))) U4 { unsigned int x, y, z, w; };
union FragCast { v16bf v; U4 q[2]; };

__device__ __forceinline__ unsigned short f2bf(float f) {
  unsigned int u = __float_as_uint(f);
  unsigned int r = (u + 0x7FFFu + ((u >> 16) & 1u)) >> 16;  // RNE
  return (unsigned short)r;
}

// Load a 16x32 bf16 fragment (A layout; B uses same layout with W-row = out-col).
// LDS is row-major [128][128] bf16 (256B row stride).
__device__ __forceinline__ v16bf ld_frag(const unsigned short* lds, int row, int k0, int lane) {
  const int kb = (lane & 16) ? 8 : 0;
  const unsigned short* p = lds + row * E_DIM + k0 + kb;
  FragCast f;
  f.q[0] = *(const U4*)(p);        // K = k0+kb   .. k0+kb+7
  f.q[1] = *(const U4*)(p + 16);   // K = k0+16+kb .. +7
  return f.v;
}

// acc[c] covers out-cols [c*16, c*16+16), rows [rowLocal, rowLocal+16)
__device__ __forceinline__ void mma_128x128(const unsigned short* Xl, const unsigned short* Wl,
                                            int rowLocal, v8f acc[8], int lane) {
#pragma unroll
  for (int k0 = 0; k0 < E_DIM; k0 += 32) {
    v16bf a = ld_frag(Xl, rowLocal + (lane & 15), k0, lane);
#pragma unroll
    for (int c = 0; c < 8; ++c) {
      v16bf b = ld_frag(Wl, c * 16 + (lane & 15), k0, lane);
      acc[c] = __builtin_amdgcn_wmma_f32_16x16x32_bf16(
          false, a, false, b, (short)0, acc[c], false, false);
    }
  }
}

// Cooperative load of a 128x128 fp32 weight slab (row stride ldw, col offset) -> bf16 LDS
__device__ __forceinline__ void load_w_lds(unsigned short* Wl, const float* W,
                                           int ldw, int colOff, int t) {
#pragma unroll 4
  for (int i = 0; i < 64; ++i) {
    int idx = i * 256 + t;
    int r = idx >> 7, c = idx & 127;
    Wl[idx] = f2bf(W[r * ldw + colOff + c]);
  }
}

__device__ __forceinline__ void load_x_lds(unsigned short* Xl, const float* X,
                                           int rowBase, int Nn, int t) {
#pragma unroll 4
  for (int i = 0; i < 64; ++i) {
    int idx = i * 256 + t;
    int r = idx >> 7, c = idx & 127;
    int gr = rowBase + r;
    if (gr >= Nn) gr = Nn - 1;  // clamp; stores are guarded
    Xl[idx] = f2bf(X[(size_t)gr * E_DIM + c]);
  }
}

// C/D layout: lane&15 = n; lanes>=16 -> M 8..15; vgpr j -> M = (lane>>4)*8 + j
__device__ __forceinline__ void store_tiles(float* out, int rowBase, int Nn,
                                            v8f acc[8], int lane, const float* bias) {
  const int n = lane & 15, hi = (lane >> 4) & 1;
#pragma unroll
  for (int c = 0; c < 8; ++c) {
    float bv = bias ? bias[c * 16 + n] : 0.f;
#pragma unroll
    for (int j = 0; j < 8; ++j) {
      int gr = rowBase + hi * 8 + j;
      if (gr < Nn) out[(size_t)gr * E_DIM + c * 16 + n] = acc[c][j] + bv;
    }
  }
}

__device__ __forceinline__ void store_tiles_lds(unsigned short* Xl, int rowLocal,
                                                v8f acc[8], int lane, const float* bias) {
  const int n = lane & 15, hi = (lane >> 4) & 1;
#pragma unroll
  for (int c = 0; c < 8; ++c) {
    float bv = bias[c * 16 + n];
#pragma unroll
    for (int j = 0; j < 8; ++j)
      Xl[(rowLocal + hi * 8 + j) * E_DIM + c * 16 + n] = f2bf(acc[c][j] + bv);
  }
}

// ---------------- Kernel 1: per-(n,k) distance MLP 1 -> 128 -> 1 ----------------
__global__ __launch_bounds__(256) void dist_kernel(
    const float* __restrict__ dmax, const float* __restrict__ W1,
    const float* __restrict__ b1, const float* __restrict__ W2,
    const float* __restrict__ b2, float* __restrict__ dout, int total) {
  int i = blockIdx.x * 256 + threadIdx.x;
  if (i >= total) return;
  float x = dmax[i];
  float acc = b2[0];
#pragma unroll 8
  for (int e = 0; e < E_DIM; ++e) {
    float h = fmaf(W1[e], x, b1[e]);        // uniform (scalar-cached) weight loads
    acc = fmaf(W2[e], fmaxf(h, 0.f), acc);
  }
  dout[i] = acc;
}

// ---- Kernel 2: feat = X@Wf^T+bf ; g = feat@Wh_a^T ; t2 = feat@Wh_b^T+bh (fused) ----
__global__ __launch_bounds__(256) void fused_feat_gt_kernel(
    const float* __restrict__ feature, const float* __restrict__ Wf,
    const float* __restrict__ bf, const float* __restrict__ Wh,
    const float* __restrict__ bh, float* __restrict__ g, float* __restrict__ t2, int Nn) {
  __shared__ unsigned short Xl[E_DIM * E_DIM];  // 32 KB
  __shared__ unsigned short Wl[E_DIM * E_DIM];  // 32 KB
  const int t = threadIdx.x, lane = t & 31, wave = t >> 5;
  const int rowBase = blockIdx.x * 128;
  v8f zero = {};
  v8f acc[8];

  // Phase 1: feat (kept on-chip, rewritten into Xl as bf16 with bias)
  load_x_lds(Xl, feature, rowBase, Nn, t);
  load_w_lds(Wl, Wf, E_DIM, 0, t);
  __syncthreads();
#pragma unroll
  for (int c = 0; c < 8; ++c) acc[c] = zero;
  mma_128x128(Xl, Wl, wave * 16, acc, lane);
  store_tiles_lds(Xl, wave * 16, acc, lane, bf);  // each wave touches only its own rows
  __syncthreads();

  // Phase 2: g = feat @ Wh[:, :128]^T
  load_w_lds(Wl, Wh, 2 * E_DIM, 0, t);
  __syncthreads();
#pragma unroll
  for (int c = 0; c < 8; ++c) acc[c] = zero;
  mma_128x128(Xl, Wl, wave * 16, acc, lane);
  store_tiles(g, rowBase + wave * 16, Nn, acc, lane, nullptr);
  __syncthreads();

  // Phase 3: t2 = feat @ Wh[:, 128:]^T + bh
  load_w_lds(Wl, Wh, 2 * E_DIM, E_DIM, t);
  __syncthreads();
#pragma unroll
  for (int c = 0; c < 8; ++c) acc[c] = zero;
  mma_128x128(Xl, Wl, wave * 16, acc, lane);
  store_tiles(t2, rowBase + wave * 16, Nn, acc, lane, bh);
}

// ---- Kernel 3: gather + fused relu/dot/mean. One wave (32 lanes x float4) per node ----
__global__ __launch_bounds__(256) void message_kernel(
    const int* __restrict__ argmax, const float* __restrict__ dpre,
    const float* __restrict__ g, const float* t2s, const float* __restrict__ Wp,
    const float* __restrict__ bp, float* __restrict__ outPos, float* sOut, int Nn) {
  const int lane = threadIdx.x & 31, wave = threadIdx.x >> 5;
  const int n = blockIdx.x * 8 + wave;
  if (n >= Nn) return;

  const float4 tv = *(const float4*)(t2s + (size_t)n * E_DIM + lane * 4);
  const float4 wp = *(const float4*)(Wp + lane * 4);
  float s0 = 0.f, s1 = 0.f, s2 = 0.f, s3 = 0.f;
  const float bpos = bp[0];

  int idx_next = argmax[n * K_ANCH];
#pragma unroll 4
  for (int k = 0; k < K_ANCH; ++k) {
    const int idx = idx_next;
    if (k + 1 < K_ANCH) {
      idx_next = argmax[n * K_ANCH + k + 1];
      __builtin_prefetch(g + (size_t)idx_next * E_DIM, 0, 0);  // global_prefetch_b8
    }
    const float dv = dpre[n * K_ANCH + k];
    const float4 gv = *(const float4*)(g + (size_t)idx * E_DIM + lane * 4);  // L2-resident row
    float m0 = fmaxf(fmaf(dv, gv.x, tv.x), 0.f);
    float m1 = fmaxf(fmaf(dv, gv.y, tv.y), 0.f);
    float m2 = fmaxf(fmaf(dv, gv.z, tv.z), 0.f);
    float m3 = fmaxf(fmaf(dv, gv.w, tv.w), 0.f);
    s0 += m0; s1 += m1; s2 += m2; s3 += m3;
    float p = fmaf(m0, wp.x, fmaf(m1, wp.y, fmaf(m2, wp.z, m3 * wp.w)));
#pragma unroll
    for (int off = 16; off > 0; off >>= 1) p += __shfl_xor(p, off, 32);  // wave32 reduce
    if (lane == 0) outPos[n * K_ANCH + k] = p + bpos;
  }
  const float inv = 1.0f / (float)K_ANCH;
  float4 sv; sv.x = s0 * inv; sv.y = s1 * inv; sv.z = s2 * inv; sv.w = s3 * inv;
  *(float4*)(sOut + (size_t)n * E_DIM + lane * 4) = sv;  // overwrites own t2 row (read already)
}

// ---- Kernel 4: out_structure = s @ Ws_a^T + edge_attr @ Ws_b^T + bs ----
__global__ __launch_bounds__(256) void structure_kernel(
    const float* __restrict__ s, const float* __restrict__ edge,
    const float* __restrict__ Ws, const float* __restrict__ bs,
    float* __restrict__ outS, int Nn) {
  __shared__ unsigned short Xl[E_DIM * E_DIM];
  __shared__ unsigned short Wl[E_DIM * E_DIM];
  const int t = threadIdx.x, lane = t & 31, wave = t >> 5;
  const int rowBase = blockIdx.x * 128;
  v8f zero = {};
  v8f acc[8];
#pragma unroll
  for (int c = 0; c < 8; ++c) acc[c] = zero;

  load_x_lds(Xl, s, rowBase, Nn, t);
  load_w_lds(Wl, Ws, 2 * E_DIM, 0, t);
  __syncthreads();
  mma_128x128(Xl, Wl, wave * 16, acc, lane);
  __syncthreads();

  load_x_lds(Xl, edge, rowBase, Nn, t);
  load_w_lds(Wl, Ws, 2 * E_DIM, E_DIM, t);
  __syncthreads();
  mma_128x128(Xl, Wl, wave * 16, acc, lane);  // accumulates on top
  store_tiles(outS, rowBase + wave * 16, Nn, acc, lane, bs);
}

extern "C" void kernel_launch(void* const* d_in, const int* in_sizes, int n_in,
                              void* d_out, int out_size, void* d_ws, size_t ws_size,
                              hipStream_t stream) {
  const float* feature = (const float*)d_in[0];
  const float* dmax    = (const float*)d_in[1];
  const int*   argmax  = (const int*)d_in[2];
  const float* edge    = (const float*)d_in[3];
  const float* W1 = (const float*)d_in[4];
  const float* b1 = (const float*)d_in[5];
  const float* W2 = (const float*)d_in[6];
  const float* b2 = (const float*)d_in[7];
  const float* Wf = (const float*)d_in[8];
  const float* bf = (const float*)d_in[9];
  const float* Wh = (const float*)d_in[10];
  const float* bh = (const float*)d_in[11];
  const float* Wp = (const float*)d_in[12];
  const float* bp = (const float*)d_in[13];
  const float* Ws = (const float*)d_in[14];
  const float* bs = (const float*)d_in[15];

  const int Nn = in_sizes[0] / E_DIM;  // 50000

  float* ws   = (float*)d_ws;
  float* d_d  = ws;                                   // [N,K]
  float* d_g  = d_d + (size_t)Nn * K_ANCH;            // [N,E]
  float* d_t2 = d_g + (size_t)Nn * E_DIM;             // [N,E], reused as s

  float* outPos    = (float*)d_out;                   // [N,K]
  float* outStruct = outPos + (size_t)Nn * K_ANCH;    // [N,E]

  dim3 blk(256);
  dist_kernel<<<(Nn * K_ANCH + 255) / 256, blk, 0, stream>>>(dmax, W1, b1, W2, b2,
                                                             d_d, Nn * K_ANCH);
  fused_feat_gt_kernel<<<(Nn + 127) / 128, blk, 0, stream>>>(feature, Wf, bf, Wh, bh,
                                                             d_g, d_t2, Nn);
  message_kernel<<<(Nn + 7) / 8, blk, 0, stream>>>(argmax, d_d, d_g, d_t2, Wp, bp,
                                                   outPos, d_t2, Nn);
  structure_kernel<<<(Nn + 127) / 128, blk, 0, stream>>>(d_t2, edge, Ws, bs,
                                                         outStruct, Nn);
}
